// TransformerBlock_86071144612025
// MI455X (gfx1250) — compile-verified, matrix operand order checked by reading, not run on hardware
//
#include <hip/hip_runtime.h>
#include <hip/hip_bf16.h>

typedef __attribute__((ext_vector_type(16))) _Float16 v16h;
typedef __attribute__((ext_vector_type(8)))  float    v8f;

// Problem constants (fixed by the reference)
#define IMG_H 128
#define IMG_W 128
#define IMG_L 16384            // H*W
#define NB    4                // batch
#define BLTOK 65536            // B * L  (divisible by BM -> no M bounds checks)
#define HIDDEN 170

__device__ __forceinline__ float siluf(float x)  { return x / (1.f + __expf(-x)); }
__device__ __forceinline__ float geluf(float x)  { return 0.5f * x * (1.f + erff(x * 0.70710678118654752f)); }

// ---------------------------------------------------------------------------
// Generic WMMA GEMM:  out[M,N] = act( bn( A[M,K] @ W[N,K]^T + bias ) ) (+resid)
// AMODE: 0 = A row-major (lda), 1 = NCHW gather (k = channel), 2 = im2col 3x3 pad1
// OUTMODE: 0 = row-major (ldo), 1 = NCHW scatter (+ optional residual)
// ACT: 0 = none, 1 = relu, 2 = softplus
// ---------------------------------------------------------------------------
struct GemmParams {
    const float* A;   long long lda;   // AMODE 0
    const float* X;   int Cin;         // AMODE 1/2 (NCHW source)
    const float* Wt;  int Kw;          // weights (N,K) row-major
    const float* bias;
    const float* bng; const float* bnb; const float* bnm; const float* bnv;
    const float* resid;                // OUTMODE 1 residual (NCHW)
    float* out;       long long ldo;   // OUTMODE 0
    int M, Nreal, Kreal;
    int Cout;                          // OUTMODE 1 channel count of out tensor
};

#define BM 128
#define BN 16
#define BK 32
#define LDSA (BK + 8)                  // 40 halves = 80B row stride (16B aligned)

union HPack16 { _Float16 h[16]; float4 q[2]; };
union HPack2  { _Float16 h[2];  float  f;    };
union Frag    { v16h v;         float4 q[2]; };

// Stage one 128x32 A tile + 32x16 B tile (stored transposed [n][k]) into LDS.
// CHECKK: only the final (partial) K-step needs k bounds handling; implemented
// branch-free via clamped addresses + select-zero (no EXEC divergence).
template<int AMODE, bool CHECKK>
__device__ __forceinline__ void stage_tiles(const GemmParams& p, int k0, int m0, int n0,
                                            int tid, _Float16 (*As)[LDSA],
                                            _Float16 (*Bs)[LDSA], bool prefetch_next) {
    // ---- A tile: 256 threads x 16 elements ----
    {
        const int row = tid >> 1;
        const int cb  = (tid & 1) * 16;
        const int gm  = m0 + row;
        const int b   = gm >> 14;
        const int hw  = gm & 16383;
        float tv[16];
        if (AMODE == 0) {
            const float* src = p.A + (long long)gm * p.lda + k0 + cb;
            if (!CHECKK) {
                const float4 q0 = *(const float4*)(src + 0);
                const float4 q1 = *(const float4*)(src + 4);
                const float4 q2 = *(const float4*)(src + 8);
                const float4 q3 = *(const float4*)(src + 12);
                tv[0]=q0.x; tv[1]=q0.y; tv[2]=q0.z; tv[3]=q0.w;
                tv[4]=q1.x; tv[5]=q1.y; tv[6]=q1.z; tv[7]=q1.w;
                tv[8]=q2.x; tv[9]=q2.y; tv[10]=q2.z; tv[11]=q2.w;
                tv[12]=q3.x; tv[13]=q3.y; tv[14]=q3.z; tv[15]=q3.w;
                if (prefetch_next) __builtin_prefetch(src + BK, 0, 0);
            } else {
                const float* rowp = p.A + (long long)gm * p.lda;
#pragma unroll
                for (int i = 0; i < 16; ++i) {
                    const int k  = k0 + cb + i;
                    const int kc = (k < p.Kreal) ? k : (p.Kreal - 1);
                    const float v = rowp[kc];
                    tv[i] = (k < p.Kreal) ? v : 0.f;
                }
            }
        } else if (AMODE == 1) {
            const float* base = p.X + (long long)b * p.Cin * IMG_L + hw;
#pragma unroll
            for (int i = 0; i < 16; ++i) {
                const int k  = k0 + cb + i;
                const int kc = CHECKK ? ((k < p.Kreal) ? k : (p.Kreal - 1)) : k;
                const float v = base[(long long)kc * IMG_L];
                tv[i] = (!CHECKK || k < p.Kreal) ? v : 0.f;
            }
        } else {
            const float* base = p.X + (long long)b * p.Cin * IMG_L;
            const int y0 = hw >> 7, x0 = hw & 127;
#pragma unroll
            for (int i = 0; i < 16; ++i) {
                const int k  = k0 + cb + i;       // K=288 is a multiple of 32: no CHECKK path
                const int c  = k / 9, r9 = k % 9;
                const int ky = r9 / 3, kx = r9 % 3;
                const int y  = y0 + ky - 1;
                const int x  = x0 + kx - 1;
                const bool inb = (y >= 0) & (y < IMG_H) & (x >= 0) & (x < IMG_W);
                const int yc = min(max(y, 0), IMG_H - 1);
                const int xc = min(max(x, 0), IMG_W - 1);
                const float v = base[(long long)c * IMG_L + yc * IMG_W + xc];
                tv[i] = inb ? v : 0.f;
            }
        }
        HPack16 hp;
#pragma unroll
        for (int i = 0; i < 16; ++i) hp.h[i] = (_Float16)tv[i];
        *(float4*)&As[row][cb]     = hp.q[0];
        *(float4*)&As[row][cb + 8] = hp.q[1];
    }
    // ---- B tile (transposed [n][k]): 256 threads x 2 elements, packed b32 store ----
    {
        const int n  = tid >> 4;            // 0..15
        const int k  = (tid & 15) * 2;      // 0,2,..,30
        const int gn = n0 + n;
        const int gnc = (gn < p.Nreal) ? gn : (p.Nreal - 1);
        const float* wrow = p.Wt + (long long)gnc * p.Kw;
        HPack2 bp;
#pragma unroll
        for (int e = 0; e < 2; ++e) {
            const int gk = k0 + k + e;
            const int gkc = CHECKK ? ((gk < p.Kreal) ? gk : (p.Kreal - 1)) : gk;
            const float v = wrow[gkc];
            const bool ok = (gn < p.Nreal) & (!CHECKK || gk < p.Kreal);
            bp.h[e] = (_Float16)(ok ? v : 0.f);
        }
        *(float*)&Bs[n][k] = bp.f;
    }
}

template<int AMODE, int OUTMODE, int ACT>
__global__ __launch_bounds__(256)
void wmma_gemm_kernel(GemmParams p) {
    __shared__ __align__(16) _Float16 As[BM][LDSA];
    __shared__ __align__(16) _Float16 Bs[BN][LDSA];   // transposed: [n][k]

    const int tid  = threadIdx.x;
    const int wave = tid >> 5;
    const int lane = tid & 31;
    const int m0 = blockIdx.x * BM;
    const int n0 = blockIdx.y * BN;

    v8f acc = {0.f, 0.f, 0.f, 0.f, 0.f, 0.f, 0.f, 0.f};

    const int mrow = wave * 16 + (lane & 15);
    const int nf   = lane & 15;
    const int kb   = (lane >> 4) * 8;

    const int kfull = p.Kreal / BK;
    const int krem  = p.Kreal - kfull * BK;

    for (int ks = 0; ks < kfull; ++ks) {
        stage_tiles<AMODE, false>(p, ks * BK, m0, n0, tid, As, Bs, ks + 1 < kfull);
        __syncthreads();
        Frag fa, fb;
        fa.q[0] = *(const float4*)&As[mrow][kb];
        fa.q[1] = *(const float4*)&As[mrow][kb + 16];
        fb.q[0] = *(const float4*)&Bs[nf][kb];
        fb.q[1] = *(const float4*)&Bs[nf][kb + 16];
        acc = __builtin_amdgcn_wmma_f32_16x16x32_f16(false, fa.v, false, fb.v,
                                                     (short)0, acc, false, false);
        __syncthreads();
    }
    if (krem) {
        stage_tiles<AMODE, true>(p, kfull * BK, m0, n0, tid, As, Bs, false);
        __syncthreads();
        Frag fa, fb;
        fa.q[0] = *(const float4*)&As[mrow][kb];
        fa.q[1] = *(const float4*)&As[mrow][kb + 16];
        fb.q[0] = *(const float4*)&Bs[nf][kb];
        fb.q[1] = *(const float4*)&Bs[nf][kb + 16];
        acc = __builtin_amdgcn_wmma_f32_16x16x32_f16(false, fa.v, false, fb.v,
                                                     (short)0, acc, false, false);
        __syncthreads();
    }

    // ---- epilogue: bias + BN + activation + scatter (+residual) ----
    const int gn = n0 + nf;
    if (gn < p.Nreal) {
#pragma unroll
        for (int j = 0; j < 8; ++j) {
            const int mf = j + ((lane >> 4) << 3);
            const int gm = m0 + wave * 16 + mf;     // M % BM == 0: always in range
            float v = acc[j];
            if (p.bias) v += p.bias[gn];
            if (p.bng)  v = (v - p.bnm[gn]) * rsqrtf(p.bnv[gn] + 1e-5f) * p.bng[gn] + p.bnb[gn];
            if (ACT == 1) v = fmaxf(v, 0.f);
            else if (ACT == 2) v = (v > 20.f) ? v : log1pf(__expf(v));
            if (OUTMODE == 0) {
                p.out[(long long)gm * p.ldo + gn] = v;
            } else {
                const int b = gm >> 14, hw = gm & 16383;
                const long long o = ((long long)b * p.Cout + gn) * IMG_L + hw;
                if (p.resid) v += p.resid[o];
                p.out[o] = v;
            }
        }
    }
}

// ---------------------------------------------------------------------------
// Per-pixel channel LayerNorm of both halves; BN1 folded into left half.
// right_ln stored NCHW == raw reshape to (B, L, 32) row-major (faithful).
// ---------------------------------------------------------------------------
__global__ void k_ln_split(const float* __restrict__ x,
                           const float* n1w, const float* n1b,
                           const float* g1, const float* b1,
                           const float* m1, const float* v1,
                           float* __restrict__ right_ln, float* __restrict__ left0) {
    const int idx = blockIdx.x * blockDim.x + threadIdx.x;
    if (idx >= BLTOK) return;
    const int b = idx >> 14, hw = idx & 16383;
    const float* xb = x + (long long)b * 64 * IMG_L + hw;
    float l[32], r[32];
    float sl = 0.f, sr = 0.f;
#pragma unroll
    for (int c = 0; c < 32; ++c) {
        l[c] = xb[(long long)c * IMG_L];
        r[c] = xb[(long long)(c + 32) * IMG_L];
        sl += l[c]; sr += r[c];
    }
    const float mul = sl * (1.f / 32.f), mur = sr * (1.f / 32.f);
    float vl = 0.f, vr = 0.f;
#pragma unroll
    for (int c = 0; c < 32; ++c) {
        const float dl = l[c] - mul, dr = r[c] - mur;
        vl += dl * dl; vr += dr * dr;
    }
    const float il = rsqrtf(vl * (1.f / 32.f) + 1e-5f);
    const float ir = rsqrtf(vr * (1.f / 32.f) + 1e-5f);
#pragma unroll
    for (int c = 0; c < 32; ++c) {
        right_ln[((long long)b * 32 + c) * IMG_L + hw] = (r[c] - mur) * ir * n1w[c] + n1b[c];
        float lv = (l[c] - mul) * il * n1w[c] + n1b[c];
        lv = (lv - m1[c]) * rsqrtf(v1[c] + 1e-5f) * g1[c] + b1[c];
        left0[((long long)b * 32 + c) * IMG_L + hw] = lv;
    }
}

// depthwise causal conv1d (k=4, left pad 3) + SiLU on xm = xz[:, 0:64]
__global__ void k_conv1d_silu(const float* __restrict__ xz, const float* w,
                              const float* bias, float* __restrict__ u) {
    const long long idx = (long long)blockIdx.x * blockDim.x + threadIdx.x;
    if (idx >= (long long)BLTOK * 64) return;
    const int d = (int)(idx & 63);
    const long long bl = idx >> 6;
    const int l = (int)(bl & 16383);
    const long long base = bl - l;  // b*L
    float s = bias[d];
#pragma unroll
    for (int j = 0; j < 4; ++j) {
        const int li = l - 3 + j;
        const int lc = (li >= 0) ? li : 0;
        const float v = xz[(base + lc) * 128 + d];
        s += (li >= 0) ? (w[d * 4 + j] * v) : 0.f;
    }
    u[bl * 64 + d] = siluf(s);
}

// selective scan: 16 lanes = 16 states per (b,d); shfl_xor tree for h.C
__global__ void k_scan(const float* __restrict__ dt, const float* __restrict__ u,
                       const float* __restrict__ xdbl, const float* __restrict__ A_log,
                       float* __restrict__ yraw) {
    const int tid  = threadIdx.x;                 // 128 threads = 8 (b,d) pairs
    const int n    = tid & 15;
    const int pair = blockIdx.x * 8 + (tid >> 4);
    const int b = pair >> 6;
    const int d = pair & 63;
    const float A = -__expf(A_log[d * 16 + n]);
    float h = 0.f;
    const long long base = (long long)b * IMG_L;
    for (int l = 0; l < IMG_L; ++l) {
        const long long bl = base + l;
        const float dtv = dt[bl * 64 + d];
        const float uv  = u[bl * 64 + d];
        const float Bv  = xdbl[bl * 48 + 2 + n];
        const float Cv  = xdbl[bl * 48 + 18 + n];
        h = __expf(dtv * A) * h + dtv * Bv * uv;
        float p = h * Cv;
        p += __shfl_xor(p, 8);
        p += __shfl_xor(p, 4);
        p += __shfl_xor(p, 2);
        p += __shfl_xor(p, 1);
        if (n == 0) yraw[bl * 64 + d] = p;
    }
}

// y = (yraw + u*D) * silu(z)
__global__ void k_ygate(const float* __restrict__ yraw, const float* __restrict__ u,
                        const float* Dp, const float* __restrict__ xz,
                        float* __restrict__ yg) {
    const long long idx = (long long)blockIdx.x * blockDim.x + threadIdx.x;
    if (idx >= (long long)BLTOK * 64) return;
    const int d = (int)(idx & 63);
    const long long bl = idx >> 6;
    const float z = xz[bl * 128 + 64 + d];
    yg[idx] = (yraw[idx] + u[idx] * Dp[d]) * siluf(z);
}

// t1 = LN2( concat(conv_left, r2) + x )
__global__ void k_ln2a(const float* __restrict__ lconv, const float* __restrict__ mout,
                       const float* __restrict__ x, const float* n2w, const float* n2b,
                       float* __restrict__ t1) {
    const int idx = blockIdx.x * blockDim.x + threadIdx.x;
    if (idx >= BLTOK) return;
    const int b = idx >> 14, hw = idx & 16383;
    float v[64];
    float s = 0.f;
#pragma unroll
    for (int c = 0; c < 32; ++c) {
        float t = lconv[((long long)b * 32 + c) * IMG_L + hw]
                + x[((long long)b * 64 + c) * IMG_L + hw];
        v[c] = t; s += t;
    }
#pragma unroll
    for (int c = 32; c < 64; ++c) {
        float t = mout[((long long)b * IMG_L + hw) * 32 + (c - 32)]
                + x[((long long)b * 64 + c) * IMG_L + hw];
        v[c] = t; s += t;
    }
    const float mu = s * (1.f / 64.f);
    float var = 0.f;
#pragma unroll
    for (int c = 0; c < 64; ++c) { const float d2 = v[c] - mu; var += d2 * d2; }
    const float inv = rsqrtf(var * (1.f / 64.f) + 1e-5f);
#pragma unroll
    for (int c = 0; c < 64; ++c)
        t1[((long long)b * 64 + c) * IMG_L + hw] = (v[c] - mu) * inv * n2w[c] + n2b[c];
}

// global average pool per (b,c)
__global__ void k_pool(const float* __restrict__ t1, float* __restrict__ pooled) {
    __shared__ float red[256];
    const int b = blockIdx.x >> 6, c = blockIdx.x & 63;
    const float* p = t1 + ((long long)b * 64 + c) * IMG_L;
    float s = 0.f;
    for (int i = threadIdx.x; i < IMG_L; i += 256) s += p[i];
    red[threadIdx.x] = s;
    __syncthreads();
    for (int off = 128; off > 0; off >>= 1) {
        if (threadIdx.x < off) red[threadIdx.x] += red[threadIdx.x + off];
        __syncthreads();
    }
    if (threadIdx.x == 0) pooled[blockIdx.x] = red[0] * (1.f / (float)IMG_L);
}

// channel-attention tiny MLP (64 -> 4 -> 64), single block
__global__ void k_ca_fc(const float* pooled, const float* w1, const float* b1,
                        const float* w2, const float* b2, float* cav) {
    __shared__ float hid[NB * 4];
    const int tid = threadIdx.x;
    if (tid < NB * 4) {
        const int b = tid >> 2, j = tid & 3;
        float s = b1[j];
        for (int k = 0; k < 64; ++k) s += w1[j * 64 + k] * pooled[b * 64 + k];
        hid[tid] = fmaxf(s, 0.f);
    }
    __syncthreads();
    const int b = tid >> 6, c = tid & 63;
    float s = b2[c];
#pragma unroll
    for (int j = 0; j < 4; ++j) s += w2[c * 4 + j] * hid[b * 4 + j];
    cav[tid] = 1.f / (1.f + __expf(-s));
}

// t2 = t1*(1+cav); lnt = LN2(t2)
__global__ void k_scale_ln(const float* __restrict__ t1, const float* __restrict__ cav,
                           const float* n2w, const float* n2b,
                           float* __restrict__ t2, float* __restrict__ lnt) {
    const int idx = blockIdx.x * blockDim.x + threadIdx.x;
    if (idx >= BLTOK) return;
    const int b = idx >> 14, hw = idx & 16383;
    float v[64];
    float s = 0.f;
#pragma unroll
    for (int c = 0; c < 64; ++c) {
        const long long o = ((long long)b * 64 + c) * IMG_L + hw;
        const float t = t1[o] * (1.f + cav[b * 64 + c]);
        t2[o] = t; v[c] = t; s += t;
    }
    const float mu = s * (1.f / 64.f);
    float var = 0.f;
#pragma unroll
    for (int c = 0; c < 64; ++c) { const float d2 = v[c] - mu; var += d2 * d2; }
    const float inv = rsqrtf(var * (1.f / 64.f) + 1e-5f);
#pragma unroll
    for (int c = 0; c < 64; ++c)
        lnt[((long long)b * 64 + c) * IMG_L + hw] = (v[c] - mu) * inv * n2w[c] + n2b[c];
}

// depthwise 3x3 pad1 + bias over 2*HIDDEN channels
__global__ void k_dwconv(const float* __restrict__ h, const float* w, const float* bias,
                         float* __restrict__ o) {
    const long long idx = (long long)blockIdx.x * blockDim.x + threadIdx.x;
    if (idx >= (long long)NB * (2 * HIDDEN) * IMG_L) return;
    const int hw = (int)(idx & 16383);
    const long long bc = idx >> 14;
    const int c = (int)(bc % (2 * HIDDEN));
    const int y = hw >> 7, x = hw & 127;
    const float* base = h + bc * IMG_L;
    float s = bias[c];
#pragma unroll
    for (int ky = 0; ky < 3; ++ky) {
#pragma unroll
        for (int kx = 0; kx < 3; ++kx) {
            const int yy = y + ky - 1, xx = x + kx - 1;
            const bool inb = (yy >= 0) & (yy < IMG_H) & (xx >= 0) & (xx < IMG_W);
            const int yc = min(max(yy, 0), IMG_H - 1);
            const int xc = min(max(xx, 0), IMG_W - 1);
            const float v = base[yc * IMG_W + xc];
            s += inb ? (w[c * 9 + ky * 3 + kx] * v) : 0.f;
        }
    }
    o[idx] = s;
}

// g = gelu(x2)*x1 + gelu(x1)*x2
__global__ void k_gate(const float* __restrict__ dwo, float* __restrict__ g) {
    const long long idx = (long long)blockIdx.x * blockDim.x + threadIdx.x;
    if (idx >= (long long)NB * HIDDEN * IMG_L) return;
    const int hw = (int)(idx & 16383);
    const long long bc = idx >> 14;
    const int b = (int)(bc / HIDDEN);
    const int c = (int)(bc % HIDDEN);
    const float x1 = dwo[((long long)b * (2 * HIDDEN) + c) * IMG_L + hw];
    const float x2 = dwo[((long long)b * (2 * HIDDEN) + c + HIDDEN) * IMG_L + hw];
    g[idx] = geluf(x2) * x1 + geluf(x1) * x2;
}

// ---------------------------------------------------------------------------
extern "C" void kernel_launch(void* const* d_in, const int* in_sizes, int n_in,
                              void* d_out, int out_size, void* d_ws, size_t ws_size,
                              hipStream_t stream) {
    (void)in_sizes; (void)n_in; (void)out_size; (void)ws_size;
    const float* x        = (const float*)d_in[0];
    const float* n1_w     = (const float*)d_in[1];
    const float* n1_b     = (const float*)d_in[2];
    const float* in_proj  = (const float*)d_in[3];
    const float* c1d_w    = (const float*)d_in[4];
    const float* c1d_b    = (const float*)d_in[5];
    const float* x_proj   = (const float*)d_in[6];
    const float* dt_w     = (const float*)d_in[7];
    const float* dt_b     = (const float*)d_in[8];
    const float* A_log    = (const float*)d_in[9];
    const float* Dp       = (const float*)d_in[10];
    const float* out_proj = (const float*)d_in[11];
    const float* c1_w     = (const float*)d_in[12];
    const float* c1_b     = (const float*)d_in[13];
    const float* c2_w     = (const float*)d_in[14];
    const float* c2_b     = (const float*)d_in[15];
    const float* c3_w     = (const float*)d_in[16];
    const float* c3_b     = (const float*)d_in[17];
    const float* ca_w1    = (const float*)d_in[18];
    const float* ca_b1    = (const float*)d_in[19];
    const float* ca_w2    = (const float*)d_in[20];
    const float* ca_b2    = (const float*)d_in[21];
    const float* n2_w     = (const float*)d_in[22];
    const float* n2_b     = (const float*)d_in[23];
    const float* pin_w    = (const float*)d_in[24];
    const float* pin_b    = (const float*)d_in[25];
    const float* dw_w     = (const float*)d_in[26];
    const float* dw_b     = (const float*)d_in[27];
    const float* pout_w   = (const float*)d_in[28];
    const float* pout_b   = (const float*)d_in[29];
    const float* bn1_g = (const float*)d_in[30];
    const float* bn1_b = (const float*)d_in[31];
    const float* bn1_m = (const float*)d_in[32];
    const float* bn1_v = (const float*)d_in[33];
    const float* bn2_g = (const float*)d_in[34];
    const float* bn2_b = (const float*)d_in[35];
    const float* bn2_m = (const float*)d_in[36];
    const float* bn2_v = (const float*)d_in[37];
    const float* bn3_g = (const float*)d_in[38];
    const float* bn3_b = (const float*)d_in[39];
    const float* bn3_m = (const float*)d_in[40];
    const float* bn3_v = (const float*)d_in[41];

    float* ws = (float*)d_ws;
    const long long BL = BLTOK;
    float* right = ws;                       // 32*BL  (later reused: mamba out)
    float* left0 = ws + 32 * BL;             // 32*BL
    float* left1 = ws + 64 * BL;             // 32*BL
    float* xz    = ws + 96 * BL;             // 128*BL (later reused: t1 / t2)
    float* ubuf  = ws + 224 * BL;            // 64*BL  (later reused: lnt)
    float* xdbl  = ws + 288 * BL;            // 48*BL
    float* dt    = ws + 336 * BL;            // 64*BL  (later reused: yg)
    float* yraw  = ws + 400 * BL;            // 64*BL
    float* hbuf  = ws + 464 * BL;            // 340*BL (later reused: gbuf)
    float* dwo   = ws + 804 * BL;            // 340*BL
    float* pooled = ws + 1144 * BL;          // 256
    float* cav    = pooled + 256;            // 256
    float* mout = right;
    float* t1   = xz;
    float* t2   = xz + 64 * BL;
    float* lnt  = ubuf;
    float* yg   = dt;
    float* gbuf = hbuf;
    float* outp = (float*)d_out;

    const dim3 blk(256);
    const int mtiles = (BLTOK + BM - 1) / BM;   // 512

    // 1) LN split (+BN1 on left)
    k_ln_split<<<dim3((BLTOK + 255) / 256), blk, 0, stream>>>(
        x, n1_w, n1_b, bn1_g, bn1_b, bn1_m, bn1_v, right, left0);

    // 2) in_proj: (BL,32) @ (128,32)^T -> xz (BL,128)
    {
        GemmParams p = {};
        p.A = right; p.lda = 32; p.Wt = in_proj; p.Kw = 32;
        p.out = xz; p.ldo = 128; p.M = BLTOK; p.Nreal = 128; p.Kreal = 32;
        wmma_gemm_kernel<0, 0, 0><<<dim3(mtiles, 8), blk, 0, stream>>>(p);
    }
    // 3) causal depthwise conv1d + SiLU -> u
    k_conv1d_silu<<<dim3((int)(((long long)BLTOK * 64 + 255) / 256)), blk, 0, stream>>>(
        xz, c1d_w, c1d_b, ubuf);
    // 4) x_proj: (BL,64) @ (34,64)^T -> xdbl (BL,48 padded)
    {
        GemmParams p = {};
        p.A = ubuf; p.lda = 64; p.Wt = x_proj; p.Kw = 64;
        p.out = xdbl; p.ldo = 48; p.M = BLTOK; p.Nreal = 34; p.Kreal = 64;
        wmma_gemm_kernel<0, 0, 0><<<dim3(mtiles, 3), blk, 0, stream>>>(p);
    }
    // 5) dt_proj + softplus: (BL,2) @ (64,2)^T -> dt (BL,64)
    {
        GemmParams p = {};
        p.A = xdbl; p.lda = 48; p.Wt = dt_w; p.Kw = 2; p.bias = dt_b;
        p.out = dt; p.ldo = 64; p.M = BLTOK; p.Nreal = 64; p.Kreal = 2;
        wmma_gemm_kernel<0, 0, 2><<<dim3(mtiles, 4), blk, 0, stream>>>(p);
    }
    // 6) selective scan
    k_scan<<<dim3(32), dim3(128), 0, stream>>>(dt, ubuf, xdbl, A_log, yraw);
    // 7) y = (yraw + u*D) * silu(z)
    k_ygate<<<dim3((int)(((long long)BLTOK * 64 + 255) / 256)), blk, 0, stream>>>(
        yraw, ubuf, Dp, xz, yg);
    // 8) out_proj: (BL,64) @ (32,64)^T -> mout (BL,32)
    {
        GemmParams p = {};
        p.A = yg; p.lda = 64; p.Wt = out_proj; p.Kw = 64;
        p.out = mout; p.ldo = 32; p.M = BLTOK; p.Nreal = 32; p.Kreal = 64;
        wmma_gemm_kernel<0, 0, 0><<<dim3(mtiles, 2), blk, 0, stream>>>(p);
    }
    // 9) conv1 3x3 (im2col GEMM) + BN2 + ReLU : left0 -> left1
    {
        GemmParams p = {};
        p.X = left0; p.Cin = 32; p.Wt = c1_w; p.Kw = 288; p.bias = c1_b;
        p.bng = bn2_g; p.bnb = bn2_b; p.bnm = bn2_m; p.bnv = bn2_v;
        p.out = left1; p.Cout = 32; p.M = BLTOK; p.Nreal = 32; p.Kreal = 288;
        wmma_gemm_kernel<2, 1, 1><<<dim3(mtiles, 2), blk, 0, stream>>>(p);
    }
    // 10) conv2 3x3 + BN3 + ReLU : left1 -> left0
    {
        GemmParams p = {};
        p.X = left1; p.Cin = 32; p.Wt = c2_w; p.Kw = 288; p.bias = c2_b;
        p.bng = bn3_g; p.bnb = bn3_b; p.bnm = bn3_m; p.bnv = bn3_v;
        p.out = left0; p.Cout = 32; p.M = BLTOK; p.Nreal = 32; p.Kreal = 288;
        wmma_gemm_kernel<2, 1, 1><<<dim3(mtiles, 2), blk, 0, stream>>>(p);
    }
    // 11) conv3 1x1 + ReLU : left0 -> left1
    {
        GemmParams p = {};
        p.X = left0; p.Cin = 32; p.Wt = c3_w; p.Kw = 32; p.bias = c3_b;
        p.out = left1; p.Cout = 32; p.M = BLTOK; p.Nreal = 32; p.Kreal = 32;
        wmma_gemm_kernel<1, 1, 1><<<dim3(mtiles, 2), blk, 0, stream>>>(p);
    }
    // 12) concat + residual + LN2 -> t1
    k_ln2a<<<dim3((BLTOK + 255) / 256), blk, 0, stream>>>(left1, mout, x, n2_w, n2_b, t1);
    // 13-14) channel attention
    k_pool<<<dim3(NB * 64), blk, 0, stream>>>(t1, pooled);
    k_ca_fc<<<dim3(1), blk, 0, stream>>>(pooled, ca_w1, ca_b1, ca_w2, ca_b2, cav);
    // 15) t2 = t1*(1+cav); lnt = LN2(t2)
    k_scale_ln<<<dim3((BLTOK + 255) / 256), blk, 0, stream>>>(t1, cav, n2_w, n2_b, t2, lnt);
    // 16) FFN pin 1x1: lnt (64ch) -> hbuf (340ch)
    {
        GemmParams p = {};
        p.X = lnt; p.Cin = 64; p.Wt = pin_w; p.Kw = 64; p.bias = pin_b;
        p.out = hbuf; p.Cout = 2 * HIDDEN; p.M = BLTOK; p.Nreal = 2 * HIDDEN; p.Kreal = 64;
        wmma_gemm_kernel<1, 1, 0><<<dim3(mtiles, (2 * HIDDEN + 15) / 16), blk, 0, stream>>>(p);
    }
    // 17) depthwise 3x3
    k_dwconv<<<dim3((int)(((long long)NB * 2 * HIDDEN * IMG_L + 255) / 256)), blk, 0, stream>>>(
        hbuf, dw_w, dw_b, dwo);
    // 18) gated GELU
    k_gate<<<dim3((int)(((long long)NB * HIDDEN * IMG_L + 255) / 256)), blk, 0, stream>>>(dwo, gbuf);
    // 19) FFN pout 1x1 + residual(t2) -> d_out
    {
        GemmParams p = {};
        p.X = gbuf; p.Cin = HIDDEN; p.Wt = pout_w; p.Kw = HIDDEN; p.bias = pout_b;
        p.resid = t2; p.out = outp; p.Cout = 64; p.M = BLTOK; p.Nreal = 64; p.Kreal = HIDDEN;
        wmma_gemm_kernel<1, 1, 0><<<dim3(mtiles, 4), blk, 0, stream>>>(p);
    }
}
